// ModifiedCLIP_10471130267936
// MI455X (gfx1250) — matrix operand order repaired and checked by reading, not compiled, not run on hardware
//
#include <hip/hip_runtime.h>

// ---------------- problem constants ----------------
#define S_   1025
#define BB   4
#define DD   768
#define HH   12
#define SS   (S_ * S_)        // 1050625
#define M1   (S_ * BB)        // 4100 rows of x / out
#define M1P  4112             // M1 padded to mult of 16
#define MP   1040             // 1025 padded to mult of 16 (mask rows)
#define KP   1056             // 1025 padded to mult of 32 (K of attn GEMM)

typedef __attribute__((ext_vector_type(16))) __bf16 v16bf;
typedef __attribute__((ext_vector_type(8)))  float  v8f;

union ABr { v16bf v; unsigned u[8]; };

__device__ inline unsigned short f2bf(float f) {
  union { float f; unsigned u; } c; c.f = f;
  unsigned r = c.u + 0x7FFFu + ((c.u >> 16) & 1u);   // round-to-nearest-even
  return (unsigned short)(r >> 16);
}
__device__ inline float safe_adj(float x) { return x + (x == 0.0f ? 1.0f : 0.0f); }

// A-tile 16x32 bf16, row-major A[lda]: lane L = row L%16;
// VGPR0..3: K = (L/16)*8 + {0..7}; VGPR4..7: K = 16 + (L/16)*8 + {0..7}
__device__ inline v16bf load_a16x32(const unsigned short* __restrict__ A, int lda,
                                    int m0, int k0, int lane) {
  const unsigned short* row = A + (size_t)(m0 + (lane & 15)) * lda + k0 + ((lane >> 4) * 8);
  ABr r;
#pragma unroll
  for (int v = 0; v < 4; ++v) r.u[v]     = *(const unsigned*)(row + 2 * v);
#pragma unroll
  for (int v = 0; v < 4; ++v) r.u[4 + v] = *(const unsigned*)(row + 16 + 2 * v);
  return r.v;
}

// B-tile 32x16 bf16 where B(k,n) = W[n][k], W row-major [ldb]:
// lane L = col L%16; VGPR0..3: K = (L/16)*8 + {0..7}; VGPR4..7: K = 16 + (L/16)*8 + {0..7}
__device__ inline v16bf load_b32x16(const unsigned short* __restrict__ W, int ldb,
                                    int n0, int k0, int lane) {
  const unsigned short* row = W + (size_t)(n0 + (lane & 15)) * ldb + k0 + ((lane >> 4) * 8);
  ABr r;
#pragma unroll
  for (int v = 0; v < 8; ++v) r.u[v] = *(const unsigned*)(row + 2 * v + (v >= 4 ? 8 : 0));
  return r.v;
}

// Pipelined 16x64 tile GEMM body: register double-buffering so global_load_b128
// for step k+1 overlaps v_wmma for step k; global_prefetch_b8 2 steps ahead.
template <int KTOT>
__device__ inline void wmma_tile16x64(const unsigned short* __restrict__ A, int lda,
                                      const unsigned short* __restrict__ W, int ldb,
                                      int m0, int n0, int lane, v8f* acc) {
  const unsigned short* arow = A + (size_t)(m0 + (lane & 15)) * lda;
  const unsigned short* wrow = W + (size_t)(n0 + (lane & 15)) * ldb;
  v16bf a  = load_a16x32(A, lda, m0, 0, lane);
  v16bf b0 = load_b32x16(W, ldb, n0,      0, lane);
  v16bf b1 = load_b32x16(W, ldb, n0 + 16, 0, lane);
  v16bf b2 = load_b32x16(W, ldb, n0 + 32, 0, lane);
  v16bf b3 = load_b32x16(W, ldb, n0 + 48, 0, lane);
  for (int k0 = 32; k0 < KTOT; k0 += 32) {
    // speculative L2 prefetch, 2 k-steps ahead (safe past end of buffer)
    __builtin_prefetch(arow + k0 + 64, 0, 1);
    __builtin_prefetch(wrow + k0 + 64, 0, 1);
    v16bf an = load_a16x32(A, lda, m0, k0, lane);
    v16bf c0 = load_b32x16(W, ldb, n0,      k0, lane);
    v16bf c1 = load_b32x16(W, ldb, n0 + 16, k0, lane);
    v16bf c2 = load_b32x16(W, ldb, n0 + 32, k0, lane);
    v16bf c3 = load_b32x16(W, ldb, n0 + 48, k0, lane);
    acc[0] = __builtin_amdgcn_wmma_f32_16x16x32_bf16(false, a, false, b0, (short)0, acc[0], false, false);
    acc[1] = __builtin_amdgcn_wmma_f32_16x16x32_bf16(false, a, false, b1, (short)0, acc[1], false, false);
    acc[2] = __builtin_amdgcn_wmma_f32_16x16x32_bf16(false, a, false, b2, (short)0, acc[2], false, false);
    acc[3] = __builtin_amdgcn_wmma_f32_16x16x32_bf16(false, a, false, b3, (short)0, acc[3], false, false);
    a = an; b0 = c0; b1 = c1; b2 = c2; b3 = c3;
  }
  acc[0] = __builtin_amdgcn_wmma_f32_16x16x32_bf16(false, a, false, b0, (short)0, acc[0], false, false);
  acc[1] = __builtin_amdgcn_wmma_f32_16x16x32_bf16(false, a, false, b1, (short)0, acc[1], false, false);
  acc[2] = __builtin_amdgcn_wmma_f32_16x16x32_bf16(false, a, false, b2, (short)0, acc[2], false, false);
  acc[3] = __builtin_amdgcn_wmma_f32_16x16x32_bf16(false, a, false, b3, (short)0, acc[3], false, false);
}

// ---------------- utility kernels ----------------
__global__ void zero_u32(unsigned* __restrict__ p, size_t n) {
  size_t i = (size_t)blockIdx.x * blockDim.x + threadIdx.x;
  size_t st = (size_t)gridDim.x * blockDim.x;
  for (; i < n; i += st) p[i] = 0u;
}

__global__ void conv_x(const float* __restrict__ x, unsigned short* __restrict__ xbf) {
  int i = blockIdx.x * 256 + threadIdx.x;
  if (i < M1 * DD) xbf[i] = f2bf(x[i]);
}

__global__ void conv_w(const float* __restrict__ ipw, const float* __restrict__ opw,
                       unsigned short* __restrict__ wv, unsigned short* __restrict__ wo) {
  int i = blockIdx.x * 256 + threadIdx.x;
  if (i < DD * DD) {
    wv[i] = f2bf(ipw[2 * DD * DD + i]);   // V-part of in_proj_w
    wo[i] = f2bf(opw[i]);
  }
}

// ---------------- mask normalization chain ----------------
__global__ void mask_colsum(const float* __restrict__ m, float* __restrict__ cs) {
  int b = blockIdx.y;
  int c = blockIdx.x * 256 + threadIdx.x;
  if (c >= S_) return;
  const float* p = m + (size_t)b * SS + c;
  float s = 0.f;
  for (int r = 0; r < S_; ++r) s += p[(size_t)r * S_];
  cs[b * S_ + c] = s;
}

__global__ void mask_rowsum1(const float* __restrict__ m, const float* __restrict__ cs,
                             float* __restrict__ rs) {
  int b = blockIdx.y, r = blockIdx.x;
  const float* row = m + (size_t)b * SS + (size_t)r * S_;
  const float* csb = cs + b * S_;
  float p = 0.f;
  for (int c = threadIdx.x; c < S_; c += 256) p += row[c] / safe_adj(csb[c]);
  __shared__ float sm[256];
  sm[threadIdx.x] = p; __syncthreads();
  for (int o = 128; o > 0; o >>= 1) {
    if ((int)threadIdx.x < o) sm[threadIdx.x] += sm[threadIdx.x + o];
    __syncthreads();
  }
  if (threadIdx.x == 0) rs[b * S_ + r] = sm[0];
}

__global__ void mask_sym(const float* __restrict__ m, const float* __restrict__ cs,
                         const float* __restrict__ rs, float* __restrict__ m3) {
  size_t idx = (size_t)blockIdx.x * 256 + threadIdx.x;
  if (idx >= (size_t)BB * SS) return;
  int c = (int)(idx % S_);
  size_t t = idx / S_;
  int r = (int)(t % S_);
  int b = (int)(t / S_);
  const float* mb  = m + (size_t)b * SS;
  const float* csb = cs + b * S_;
  const float* rsb = rs + b * S_;
  float v1 = mb[(size_t)r * S_ + c] / (safe_adj(csb[c]) * safe_adj(rsb[r]));
  float v2 = mb[(size_t)c * S_ + r] / (safe_adj(csb[r]) * safe_adj(rsb[c]));
  m3[idx] = 0.5f * (v1 + v2);
}

__global__ void mask_colmean(const float* __restrict__ m3, float* __restrict__ cm) {
  int b = blockIdx.y;
  int c = blockIdx.x * 256 + threadIdx.x;
  if (c >= S_) return;
  const float* p = m3 + (size_t)b * SS + c;
  float s = 0.f;
  for (int r = 0; r < S_; ++r) s += p[(size_t)r * S_];
  cm[b * S_ + c] = s * (1.0f / (float)S_);
}

__global__ void mask_rowsum5(const float* __restrict__ m3, const float* __restrict__ cm,
                             float* __restrict__ rs5) {
  int b = blockIdx.y, r = blockIdx.x;
  const float* row = m3 + (size_t)b * SS + (size_t)r * S_;
  const float* cmb = cm + b * S_;
  float p = 0.f;
  for (int c = threadIdx.x; c < S_; c += 256) {
    float v = row[c] - cmb[c];
    p += v > 0.f ? v : 0.f;
  }
  __shared__ float sm[256];
  sm[threadIdx.x] = p; __syncthreads();
  for (int o = 128; o > 0; o >>= 1) {
    if ((int)threadIdx.x < o) sm[threadIdx.x] += sm[threadIdx.x + o];
    __syncthreads();
  }
  if (threadIdx.x == 0) rs5[b * S_ + r] = sm[0];
}

__global__ void mask_final(const float* __restrict__ m3, const float* __restrict__ cm,
                           const float* __restrict__ rs5,
                           unsigned short* __restrict__ mfin, float* __restrict__ out2) {
  size_t idx = (size_t)blockIdx.x * 256 + threadIdx.x;
  if (idx >= (size_t)BB * SS) return;
  int c = (int)(idx % S_);
  size_t t = idx / S_;
  int r = (int)(t % S_);
  int b = (int)(t / S_);
  float v = m3[idx] - cm[b * S_ + c];
  v = v > 0.f ? v : 0.f;
  v = v / safe_adj(rs5[b * S_ + r]);
  mfin[(size_t)b * MP * KP + (size_t)r * KP + c] = f2bf(v);
  float* o = out2 + (size_t)b * HH * SS + (size_t)r * S_ + c;
#pragma unroll
  for (int h = 0; h < HH; ++h) o[(size_t)h * SS] = v;
}

// ---------------- WMMA GEMMs (1 wave / block, 16x64 tile / wave) ----------------
// GEMM1: V = x @ Wv^T + bv ; store transposed bf16  Vt[b][d][s_pad]
__global__ void __launch_bounds__(32) gemm1_proj_v(
    const unsigned short* __restrict__ xbf, const unsigned short* __restrict__ wv,
    const float* __restrict__ inb, unsigned short* __restrict__ vt) {
  const int lane = threadIdx.x;
  const int m0 = blockIdx.x * 16, n0 = blockIdx.y * 64;
  v8f acc[4] = {};
  wmma_tile16x64<DD>(xbf, DD, wv, DD, m0, n0, lane, acc);
  const int nl = lane & 15, mh = (lane >> 4) * 8;
#pragma unroll
  for (int t = 0; t < 4; ++t) {
    int d = n0 + 16 * t + nl;
    float bias = inb[2 * DD + d];
#pragma unroll
    for (int g = 0; g < 8; ++g) {
      int r = m0 + mh + g;
      if (r < M1) {
        int s = r >> 2, b = r & 3;
        vt[(size_t)(b * DD + d) * KP + s] = f2bf(acc[t][g] + bias);
      }
    }
  }
}

// GEMM3: Y[s,b,:] = m_final[b] @ V[:,b,:]  (A = mfin[b], B(k,n) = Vt[b][n][k])
__global__ void __launch_bounds__(32) gemm3_attn(
    const unsigned short* __restrict__ mfin, const unsigned short* __restrict__ vt,
    unsigned short* __restrict__ ypre) {
  const int lane = threadIdx.x;
  const int m0 = blockIdx.x * 16, n0 = blockIdx.y * 64, b = blockIdx.z;
  const unsigned short* A = mfin + (size_t)b * MP * KP;
  const unsigned short* W = vt + (size_t)b * DD * KP;
  v8f acc[4] = {};
  wmma_tile16x64<KP>(A, KP, W, KP, m0, n0, lane, acc);
  const int nl = lane & 15, mh = (lane >> 4) * 8;
#pragma unroll
  for (int t = 0; t < 4; ++t) {
    int d = n0 + 16 * t + nl;
#pragma unroll
    for (int g = 0; g < 8; ++g) {
      int s = m0 + mh + g;
      if (s < S_) ypre[(size_t)(s * BB + b) * DD + d] = f2bf(acc[t][g]);
    }
  }
}

// GEMM4: out = Y @ Wout^T + ob  (fp32 store into d_out region 1)
__global__ void __launch_bounds__(32) gemm4_outproj(
    const unsigned short* __restrict__ ypre, const unsigned short* __restrict__ wo,
    const float* __restrict__ ob, float* __restrict__ out) {
  const int lane = threadIdx.x;
  const int m0 = blockIdx.x * 16, n0 = blockIdx.y * 64;
  v8f acc[4] = {};
  wmma_tile16x64<DD>(ypre, DD, wo, DD, m0, n0, lane, acc);
  const int nl = lane & 15, mh = (lane >> 4) * 8;
#pragma unroll
  for (int t = 0; t < 4; ++t) {
    int d = n0 + 16 * t + nl;
    float bias = ob[d];
#pragma unroll
    for (int g = 0; g < 8; ++g) {
      int r = m0 + mh + g;
      if (r < M1) out[(size_t)r * DD + d] = acc[t][g] + bias;
    }
  }
}

// ---------------- host launch ----------------
extern "C" void kernel_launch(void* const* d_in, const int* in_sizes, int n_in,
                              void* d_out, int out_size, void* d_ws, size_t ws_size,
                              hipStream_t stream) {
  const float* x   = (const float*)d_in[0];
  const float* am  = (const float*)d_in[1];
  const float* ipw = (const float*)d_in[2];
  const float* ipb = (const float*)d_in[3];
  const float* opw = (const float*)d_in[4];
  const float* opb = (const float*)d_in[5];
  float* out1 = (float*)d_out;                       // [S,B,D]
  float* out2 = out1 + (size_t)S_ * BB * DD;         // [B*H,S,S]

  // workspace layout (all offsets 16B-aligned)
  char* ws = (char*)d_ws;
  size_t off = 0;
  unsigned short* xbf  = (unsigned short*)(ws + off); off += (size_t)M1P * DD * 2;      // 6,316,032
  unsigned short* wvbf = (unsigned short*)(ws + off); off += (size_t)DD * DD * 2;       // +1,179,648
  unsigned short* wobf = (unsigned short*)(ws + off); off += (size_t)DD * DD * 2;
  unsigned short* mfin = (unsigned short*)(ws + off); off += (size_t)BB * MP * KP * 2;  // +8,785,920
  unsigned short* vt   = (unsigned short*)(ws + off); off += (size_t)BB * DD * KP * 2;  // +6,488,064
  unsigned short* ypre = (unsigned short*)(ws + off); off += (size_t)M1P * DD * 2;
  size_t bf16_bytes = off;                                                              // 30,265,344
  float* m3  = (float*)(ws + off); off += (size_t)BB * SS * 4;                          // +16,810,000
  float* cs  = (float*)(ws + off); off += (size_t)BB * S_ * 4 + 112;
  float* rs1 = (float*)(ws + off); off += (size_t)BB * S_ * 4 + 112;
  float* cm  = (float*)(ws + off); off += (size_t)BB * S_ * 4 + 112;
  float* rs5 = (float*)(ws + off); off += (size_t)BB * S_ * 4 + 112;
  (void)ws_size; (void)in_sizes; (void)n_in; (void)out_size;

  // 1) zero bf16 region (covers all zero-padding for WMMA tiles)
  zero_u32<<<2048, 256, 0, stream>>>((unsigned*)ws, bf16_bytes / 4);
  // 2) fp32 -> bf16 conversions
  conv_x<<<(M1 * DD + 255) / 256, 256, 0, stream>>>(x, xbf);
  conv_w<<<(DD * DD + 255) / 256, 256, 0, stream>>>(ipw, opw, wvbf, wobf);
  // 3) mask normalization chain
  dim3 gcol((S_ + 255) / 256, BB);
  mask_colsum<<<gcol, 256, 0, stream>>>(am, cs);
  mask_rowsum1<<<dim3(S_, BB), 256, 0, stream>>>(am, cs, rs1);
  mask_sym<<<(unsigned)(((size_t)BB * SS + 255) / 256), 256, 0, stream>>>(am, cs, rs1, m3);
  mask_colmean<<<gcol, 256, 0, stream>>>(m3, cm);
  mask_rowsum5<<<dim3(S_, BB), 256, 0, stream>>>(m3, cm, rs5);
  mask_final<<<(unsigned)(((size_t)BB * SS + 255) / 256), 256, 0, stream>>>(m3, cm, rs5, mfin, out2);
  // 4) WMMA GEMMs
  gemm1_proj_v<<<dim3(M1P / 16, DD / 64), 32, 0, stream>>>(xbf, wvbf, ipb, vt);
  gemm3_attn  <<<dim3(MP / 16, DD / 64, BB), 32, 0, stream>>>(mfin, vt, ypre);
  gemm4_outproj<<<dim3(M1P / 16, DD / 64), 32, 0, stream>>>(ypre, wobf, opb, out1);
}
